// MoE_52802327937614
// MI455X (gfx1250) — matrix operand order repaired and checked by reading, not compiled, not run on hardware
//
#include <hip/hip_runtime.h>
#include <hip/hip_bf16.h>

// ---------------- types ----------------
typedef __attribute__((ext_vector_type(16))) __bf16 v16bf;
typedef __attribute__((ext_vector_type(8)))  __bf16 v8bf;
typedef __attribute__((ext_vector_type(8)))  float  v8f;

union V16 { v16bf v; v8bf h[2]; };

#define DIM  1024
#define HID  2048
#define NE   8
#define TOPK 2
#define CAP  2048            // max rows per expert (top-2 of 8 -> <= T)

// ---------------- 0: init counters ----------------
__global__ void moe_init(int* __restrict__ cnt, int* __restrict__ fill) {
    int t = threadIdx.x;
    if (t < NE) { cnt[t] = 0; fill[t] = 0; }
}

// ---------------- 1: router (one wave per token) ----------------
__global__ __launch_bounds__(256)
void moe_router(const float* __restrict__ x, const float* __restrict__ rw,
                int* __restrict__ eid, float* __restrict__ scr,
                int* __restrict__ cnt, int T) {
    int tid  = threadIdx.x;
    int wave = tid >> 5;
    int lane = tid & 31;
    int t = blockIdx.x * 8 + wave;
    if (t >= T) return;

    const float* xr = x + (size_t)t * DIM;
    float acc[NE];
#pragma unroll
    for (int e = 0; e < NE; ++e) acc[e] = 0.f;

    for (int i = lane; i < DIM; i += 32) {
        float xv = xr[i];
        const float4* w4 = (const float4*)(rw + (size_t)i * NE);
        float4 wa = w4[0], wb = w4[1];
        acc[0] += xv * wa.x; acc[1] += xv * wa.y;
        acc[2] += xv * wa.z; acc[3] += xv * wa.w;
        acc[4] += xv * wb.x; acc[5] += xv * wb.y;
        acc[6] += xv * wb.z; acc[7] += xv * wb.w;
    }
#pragma unroll
    for (int m = 16; m >= 1; m >>= 1)
#pragma unroll
        for (int e = 0; e < NE; ++e)
            acc[e] += __shfl_xor(acc[e], m, 32);

    if (lane == 0) {
        float mx = acc[0];
#pragma unroll
        for (int e = 1; e < NE; ++e) mx = fmaxf(mx, acc[e]);
        float p[NE], sum = 0.f;
#pragma unroll
        for (int e = 0; e < NE; ++e) { p[e] = __expf(acc[e] - mx); sum += p[e]; }
        float inv = __builtin_amdgcn_rcpf(sum);

        int i1 = 0; float b1 = p[0];
#pragma unroll
        for (int e = 1; e < NE; ++e) if (p[e] > b1) { b1 = p[e]; i1 = e; }
        int i2 = -1; float b2 = -1.f;
#pragma unroll
        for (int e = 0; e < NE; ++e) if (e != i1 && p[e] > b2) { b2 = p[e]; i2 = e; }

        eid[t * 2 + 0] = i1; scr[t * 2 + 0] = b1 * inv;
        eid[t * 2 + 1] = i2; scr[t * 2 + 1] = b2 * inv;
        atomicAdd(&cnt[i1], 1);
        atomicAdd(&cnt[i2], 1);
    }
}

// ---------------- 2: prefix sum of expert counts ----------------
__global__ void moe_prefix(const int* __restrict__ cnt, int* __restrict__ base) {
    if (threadIdx.x == 0) {
        int acc = 0;
#pragma unroll
        for (int e = 0; e < NE; ++e) { base[e] = acc; acc += cnt[e]; }
    }
}

// ---------------- 3: dispatch (gather + pre-scale -> bf16 xbuf) ----------------
__global__ __launch_bounds__(256)
void moe_dispatch(const float* __restrict__ x, const int* __restrict__ eid,
                  const float* __restrict__ scr, const int* __restrict__ base,
                  int* __restrict__ fill, int* __restrict__ slot_of,
                  __bf16* __restrict__ xbuf) {
    int n = blockIdx.x;                 // routed assignment index [0, T*2)
    int t = n >> 1;
    int e = eid[n];
    float s = scr[n];

    __shared__ int s_slot;
    if (threadIdx.x == 0) {
        int pos = atomicAdd(&fill[e], 1);
        int slot = base[e] + pos;
        s_slot = slot;
        slot_of[n] = slot;
    }
    __syncthreads();
    int slot = s_slot;

    const float* xr = x + (size_t)t * DIM;
    __bf16* xo = xbuf + (size_t)slot * DIM;
    for (int d = threadIdx.x; d < DIM; d += 256)
        xo[d] = (__bf16)(xr[d] * s);
}

// ---------------- WMMA fragment helpers (ISA 7.12.2 layouts) ----------------
// A 16x32 bf16: lane L -> row m=L%16; halves k0=(L>>4)*8 and k0+16 (8 contiguous each)
// B 32x16 bf16 staged n-major in LDS: lane L -> col n=L%16; halves k0=(L>>4)*8 and k0+16 (8 contiguous each)
// C/D 16x16 f32: VGPR i -> row i + (L>>4)*8, col L%16

#define LDA 40     // LDS stride (elements): 80B rows, 16B-aligned fragment chunks
#define LDB 40

#define A_TILE (32 * LDA)     // elements per A k-tile
#define B_TILE (256 * LDB)    // elements per B k-tile

__device__ __forceinline__ v16bf loadA_frag(const __bf16* sA, int lane, int mi) {
    int m  = mi * 16 + (lane & 15);
    int k0 = (lane >> 4) * 8;
    V16 f;
    f.h[0] = *(const v8bf*)(sA + m * LDA + k0);
    f.h[1] = *(const v8bf*)(sA + m * LDA + k0 + 16);
    return f.v;
}
__device__ __forceinline__ v16bf loadB_frag(const __bf16* sB, int lane, int ncol) {
    int n  = ncol + (lane & 15);
    int k0 = (lane >> 4) * 8;
    V16 f;
    f.h[0] = *(const v8bf*)(sB + n * LDB + k0);
    f.h[1] = *(const v8bf*)(sB + n * LDB + k0 + 16);
    return f.v;
}

#define WMMA_BF16(A, B, C) __builtin_amdgcn_wmma_f32_16x16x32_bf16( \
        false, (A), false, (B), (short)0, (C), false, false)

// ---------------- 4: expert GEMM-1 (fused w1 & w3, SwiGLU) ----------------
// Tile: M=32 x N=256, K=DIM in 32-steps, double-buffered LDS (dynamic, 85KB).
// 8 waves; each wave: full M (2 A-frags) x 32 cols (2 B-frags per matrix) -> 8 wmma/step.
__global__ __launch_bounds__(256)
void moe_gemm1(const __bf16* __restrict__ xbuf,
               const float* __restrict__ w1, const float* __restrict__ w3,
               __bf16* __restrict__ hbuf,
               const int* __restrict__ cnt, const int* __restrict__ base) {
    int e = blockIdx.z;
    int c = cnt[e];
    int row0 = blockIdx.y * 32;
    if (row0 >= c) return;
    int n0 = blockIdx.x * 256;
    int b  = base[e];

    extern __shared__ __align__(16) __bf16 smem[];
    __bf16* sA  = smem;                    // 2 * A_TILE
    __bf16* sB1 = sA  + 2 * A_TILE;        // 2 * B_TILE
    __bf16* sB3 = sB1 + 2 * B_TILE;        // 2 * B_TILE

    int tid  = threadIdx.x;
    int wave = tid >> 5, lane = tid & 31;
    int cb   = wave * 32;                  // this wave's column base within tile

    v8f acc1[2][2], acc3[2][2];
#pragma unroll
    for (int mi = 0; mi < 2; ++mi)
#pragma unroll
        for (int ni = 0; ni < 2; ++ni)
#pragma unroll
            for (int i = 0; i < 8; ++i) { acc1[mi][ni][i] = 0.f; acc3[mi][ni][i] = 0.f; }

    // A loader mapping: 8 threads x 4 bf16 per row
    int lr = tid >> 3;
    int lc = (tid & 7) * 4;
    int arow = row0 + lr; if (arow > c - 1) arow = c - 1;
    const __bf16* aP = xbuf + (size_t)(b + arow) * DIM + lc;
    const float* w1P = w1 + (size_t)e * DIM * HID + n0;
    const float* w3P = w3 + (size_t)e * DIM * HID + n0;

    auto load_tile = [&](int buf, int k0) {
        __bf16* dA  = sA  + buf * A_TILE;
        __bf16* dB1 = sB1 + buf * B_TILE;
        __bf16* dB3 = sB3 + buf * B_TILE;
        *(uint2*)(dA + lr * LDA + lc) = *(const uint2*)(aP + k0);
#pragma unroll 8
        for (int kk = 0; kk < 32; ++kk) {
            float v1 = w1P[(size_t)(k0 + kk) * HID + tid];
            float v3 = w3P[(size_t)(k0 + kk) * HID + tid];
            dB1[tid * LDB + kk] = (__bf16)v1;
            dB3[tid * LDB + kk] = (__bf16)v3;
        }
    };

    const int NSTEP = DIM / 32;
    load_tile(0, 0);
    __syncthreads();

    for (int s = 0; s < NSTEP; ++s) {
        int cur = s & 1;
        if (s + 1 < NSTEP) load_tile(cur ^ 1, (s + 1) * 32);

        const __bf16* cA  = sA  + cur * A_TILE;
        const __bf16* cB1 = sB1 + cur * B_TILE;
        const __bf16* cB3 = sB3 + cur * B_TILE;

        // load ALL fragments first so ds_loads pipeline behind one wait
        v16bf aF[2], bF1[2], bF3[2];
#pragma unroll
        for (int mi = 0; mi < 2; ++mi) aF[mi] = loadA_frag(cA, lane, mi);
#pragma unroll
        for (int ni = 0; ni < 2; ++ni) {
            bF1[ni] = loadB_frag(cB1, lane, cb + ni * 16);
            bF3[ni] = loadB_frag(cB3, lane, cb + ni * 16);
        }
#pragma unroll
        for (int mi = 0; mi < 2; ++mi)
#pragma unroll
            for (int ni = 0; ni < 2; ++ni) {
                acc1[mi][ni] = WMMA_BF16(aF[mi], bF1[ni], acc1[mi][ni]);
                acc3[mi][ni] = WMMA_BF16(aF[mi], bF3[ni], acc3[mi][ni]);
            }
        __syncthreads();
    }

    // SwiGLU epilogue -> bf16 hbuf (fast sigmoid via v_rcp_f32)
#pragma unroll
    for (int mi = 0; mi < 2; ++mi)
#pragma unroll
        for (int ni = 0; ni < 2; ++ni) {
            int col = n0 + cb + ni * 16 + (lane & 15);
#pragma unroll
            for (int i = 0; i < 8; ++i) {
                int m  = mi * 16 + (lane >> 4) * 8 + i;
                int gr = row0 + m;
                if (gr < c) {
                    float a = acc1[mi][ni][i];
                    float g = acc3[mi][ni][i];
                    float h = a * __builtin_amdgcn_rcpf(1.f + __expf(-a)) * g;
                    hbuf[(size_t)(b + gr) * HID + col] = (__bf16)h;
                }
            }
        }
}

// ---------------- 5: expert GEMM-2 (H @ w2 -> f32 ybuf) ----------------
// Tile: M=32 x N=256, K=HID in 32-steps, double-buffered static LDS (46KB).
__global__ __launch_bounds__(256)
void moe_gemm2(const __bf16* __restrict__ hbuf, const float* __restrict__ w2,
               float* __restrict__ ybuf,
               const int* __restrict__ cnt, const int* __restrict__ base) {
    int e = blockIdx.z;
    int c = cnt[e];
    int row0 = blockIdx.y * 32;
    if (row0 >= c) return;
    int n0 = blockIdx.x * 256;
    int b  = base[e];

    __shared__ __align__(16) __bf16 sA[2 * A_TILE];
    __shared__ __align__(16) __bf16 sB[2 * B_TILE];

    int tid  = threadIdx.x;
    int wave = tid >> 5, lane = tid & 31;
    int cb   = wave * 32;

    v8f acc[2][2];
#pragma unroll
    for (int mi = 0; mi < 2; ++mi)
#pragma unroll
        for (int ni = 0; ni < 2; ++ni)
#pragma unroll
            for (int i = 0; i < 8; ++i) acc[mi][ni][i] = 0.f;

    int lr = tid >> 3;
    int lc = (tid & 7) * 4;
    int arow = row0 + lr; if (arow > c - 1) arow = c - 1;
    const __bf16* aP = hbuf + (size_t)(b + arow) * HID + lc;
    const float* w2P = w2 + (size_t)e * HID * DIM + n0;

    auto load_tile = [&](int buf, int k0) {
        __bf16* dA = sA + buf * A_TILE;
        __bf16* dB = sB + buf * B_TILE;
        *(uint2*)(dA + lr * LDA + lc) = *(const uint2*)(aP + k0);
#pragma unroll 8
        for (int kk = 0; kk < 32; ++kk) {
            float v = w2P[(size_t)(k0 + kk) * DIM + tid];
            dB[tid * LDB + kk] = (__bf16)v;
        }
    };

    const int NSTEP = HID / 32;
    load_tile(0, 0);
    __syncthreads();

    for (int s = 0; s < NSTEP; ++s) {
        int cur = s & 1;
        if (s + 1 < NSTEP) load_tile(cur ^ 1, (s + 1) * 32);

        const __bf16* cA = sA + cur * A_TILE;
        const __bf16* cB = sB + cur * B_TILE;

        v16bf aF[2], bF[2];
#pragma unroll
        for (int mi = 0; mi < 2; ++mi) aF[mi] = loadA_frag(cA, lane, mi);
#pragma unroll
        for (int ni = 0; ni < 2; ++ni) bF[ni] = loadB_frag(cB, lane, cb + ni * 16);
#pragma unroll
        for (int mi = 0; mi < 2; ++mi)
#pragma unroll
            for (int ni = 0; ni < 2; ++ni)
                acc[mi][ni] = WMMA_BF16(aF[mi], bF[ni], acc[mi][ni]);
        __syncthreads();
    }

#pragma unroll
    for (int mi = 0; mi < 2; ++mi)
#pragma unroll
        for (int ni = 0; ni < 2; ++ni) {
            int col = n0 + cb + ni * 16 + (lane & 15);
#pragma unroll
            for (int i = 0; i < 8; ++i) {
                int m  = mi * 16 + (lane >> 4) * 8 + i;
                int gr = row0 + m;
                if (gr < c)
                    ybuf[(size_t)(b + gr) * DIM + col] = acc[mi][ni][i];
            }
        }
}

// ---------------- 6: combine (deterministic, no atomics) ----------------
__global__ __launch_bounds__(256)
void moe_combine(const float* __restrict__ ybuf, const int* __restrict__ slot_of,
                 float* __restrict__ out) {
    int t = blockIdx.x;
    int s0 = slot_of[t * 2 + 0];
    int s1 = slot_of[t * 2 + 1];
    const float* y0 = ybuf + (size_t)s0 * DIM;
    const float* y1 = ybuf + (size_t)s1 * DIM;
    float* o = out + (size_t)t * DIM;
    for (int d = threadIdx.x; d < DIM; d += 256)
        o[d] = y0[d] + y1[d];
}

// ---------------- host launcher ----------------
extern "C" void kernel_launch(void* const* d_in, const int* in_sizes, int n_in,
                              void* d_out, int out_size, void* d_ws, size_t ws_size,
                              hipStream_t stream) {
    const float* x  = (const float*)d_in[0];
    const float* rw = (const float*)d_in[1];
    const float* w1 = (const float*)d_in[2];
    const float* w2 = (const float*)d_in[3];
    const float* w3 = (const float*)d_in[4];
    float* out = (float*)d_out;

    const int T = in_sizes[0] / DIM;   // tokens
    const int N = T * TOPK;            // routed rows

    // workspace carve-out (256B aligned)
    char* wsb = (char*)d_ws;
    size_t off = 0;
    auto alloc = [&](size_t bytes) -> void* {
        off = (off + 255) & ~(size_t)255;
        void* p = wsb + off;
        off += bytes;
        return p;
    };
    int*    cnt     = (int*)   alloc(NE * sizeof(int));
    int*    base    = (int*)   alloc(NE * sizeof(int));
    int*    fill    = (int*)   alloc(NE * sizeof(int));
    int*    eid     = (int*)   alloc((size_t)N * sizeof(int));
    float*  scr     = (float*) alloc((size_t)N * sizeof(float));
    int*    slot_of = (int*)   alloc((size_t)N * sizeof(int));
    __bf16* xbuf    = (__bf16*)alloc((size_t)N * DIM * sizeof(__bf16));
    __bf16* hbuf    = (__bf16*)alloc((size_t)N * HID * sizeof(__bf16));
    float*  ybuf    = (float*) alloc((size_t)N * DIM * sizeof(float));
    (void)ws_size; (void)n_in; (void)out_size;

    const size_t g1_lds = (size_t)(2 * A_TILE + 4 * B_TILE) * sizeof(__bf16); // ~85KB

    moe_init<<<1, 32, 0, stream>>>(cnt, fill);
    moe_router<<<(T + 7) / 8, 256, 0, stream>>>(x, rw, eid, scr, cnt, T);
    moe_prefix<<<1, 1, 0, stream>>>(cnt, base);
    moe_dispatch<<<N, 256, 0, stream>>>(x, eid, scr, base, fill, slot_of, xbuf);
    moe_gemm1<<<dim3(HID / 256, CAP / 32, NE), 256, g1_lds, stream>>>(xbuf, w1, w3, hbuf, cnt, base);
    moe_gemm2<<<dim3(DIM / 256, CAP / 32, NE), 256, 0, stream>>>(hbuf, w2, ybuf, cnt, base);
    moe_combine<<<T, 256, 0, stream>>>(ybuf, slot_of, out);
}
